// EGraphNet_27298812133380
// MI455X (gfx1250) — matrix operand order repaired and checked by reading, not compiled, not run on hardware
//
#include <hip/hip_runtime.h>
#include <stdint.h>

typedef __attribute__((ext_vector_type(16))) _Float16 v16h;
typedef __attribute__((ext_vector_type(8)))  _Float16 v8h;
typedef __attribute__((ext_vector_type(8)))  float    v8f;

#define NNODE 100000
#define NCLS  50000
#define NEDGE 400000
#define NDIM  43
#define CDIM  2
#define HID   128
#define NH    4
#define SLOPE 0.2f
#define LNEPS 1e-5f
#define ORD_NEG_INF 0x007FFFFFu   // order-preserving encoding of -inf

// ---------- helpers ----------
__device__ __forceinline__ unsigned f2ord(float f) {
  unsigned u = __float_as_uint(f);
  return (u & 0x80000000u) ? ~u : (u | 0x80000000u);
}
__device__ __forceinline__ float ord2f(unsigned u) {
  return __uint_as_float((u & 0x80000000u) ? (u & 0x7FFFFFFFu) : ~u);
}
__device__ __forceinline__ void atomAddF(float* p, float v) {
  __hip_atomic_fetch_add(p, v, __ATOMIC_RELAXED, __HIP_MEMORY_SCOPE_AGENT);
}

// ---------- generic fill ----------
__global__ void fill_u32(unsigned* __restrict__ p, unsigned v, long n) {
  long i = (long)blockIdx.x * blockDim.x + threadIdx.x;
  if (i < n) p[i] = v;
}

// ---------- init smax (ordered -inf) + den (0) in one launch ----------
__global__ void init_softmax_bufs(unsigned* __restrict__ smax, float* __restrict__ den, long n) {
  long i = (long)blockIdx.x * blockDim.x + threadIdx.x;
  if (i < n) { smax[i] = ORD_NEG_INF; den[i] = 0.f; }
}

// ---------- input projections (irregular K, cheap) ----------
template <int K>
__global__ void proj_kernel(const float* __restrict__ x, const float* __restrict__ w,
                            const float* __restrict__ b, float* __restrict__ y,
                            _Float16* __restrict__ y16, int nrows) {
  int i = blockIdx.x * blockDim.x + threadIdx.x;
  if (i >= nrows * HID) return;
  int row = i >> 7, col = i & 127;
  float acc = b[col];
  const float* xr = x + (long)row * K;
#pragma unroll
  for (int k = 0; k < K; ++k) acc += xr[k] * w[k * HID + col];
  y[i] = acc;
  y16[i] = (_Float16)acc;
}

// ---------- pack two 128x128 f32 weights into WMMA B-fragment order (f16) ----------
// B-fragment (32x16 f16, K x N): lane l -> col n=l&15, half=l>>4;
// vgpr v holds K = kstep*32 + half*8 + 2v (+1) + (v>=4 ? 8 : 0). Packed so each lane reads 32B contiguous.
__global__ void pack_w(const float* __restrict__ wl, const float* __restrict__ wr,
                       _Float16* __restrict__ wpk) {
  int t = blockIdx.x * blockDim.x + threadIdx.x;
  if (t >= 2048) return;
  const float* w = (t < 1024) ? wl : wr;
  _Float16* dst = wpk + ((t < 1024) ? 0 : 16384);
  int u = t & 1023;
  int lane = u & 31, ks = (u >> 5) & 3, ct = u >> 7;
  int n = lane & 15, half = lane >> 4;
  _Float16* o = dst + (size_t)u * 16;
#pragma unroll
  for (int v = 0; v < 8; ++v) {
    int k = ks * 32 + half * 8 + 2 * v + (v >= 4 ? 8 : 0);
    o[2 * v]     = (_Float16)w[(size_t)k * HID + ct * 16 + n];
    o[2 * v + 1] = (_Float16)w[(size_t)(k + 1) * HID + ct * 16 + n];
  }
}

// ---------- Y[M x 128] = A16[M x 128] @ W + bias  via v_wmma_f32_16x16x32_f16 ----------
// block = 256 threads = 8 waves; block covers 16 rows; wave ct covers cols [ct*16, ct*16+16)
// REQUIRES: M % 16 == 0 (true for all launches here) -> no tail predication, straight-line stores.
__global__ void __launch_bounds__(256)
gemm128_wmma(const _Float16* __restrict__ A, const _Float16* __restrict__ Wpk,
             const float* __restrict__ bias, float* __restrict__ Y, int M) {
  int lane = threadIdx.x & 31;
  int ct = threadIdx.x >> 5;
  int row0 = blockIdx.x * 16;
  int n = lane & 15, half = lane >> 4;
  float bv = bias[ct * 16 + n];
  v8f acc;
#pragma unroll
  for (int r = 0; r < 8; ++r) acc[r] = bv;
  const _Float16* arow = A + (size_t)(row0 + n) * HID + half * 8;
  const _Float16* wb = Wpk + ((size_t)ct * 128 + lane) * 16;
#pragma unroll
  for (int ks = 0; ks < 4; ++ks) {
    v16h a, b;
    *(v8h*)&a       = *(const v8h*)(arow + ks * 32);        // K group 0 (8 f16, 16B)
    *((v8h*)&a + 1) = *(const v8h*)(arow + ks * 32 + 16);   // K group 1
    b = *(const v16h*)(wb + (size_t)ks * 512);              // packed, 32B coalesced
    acc = __builtin_amdgcn_wmma_f32_16x16x32_f16(false, a, false, b, (short)0, acc,
                                                 false, false);
  }
  // lane writes rows row0+half*8 .. +7, col ct*16+n : 8 stores at immediate offsets r*HID
  float* yp = Y + (size_t)(row0 + half * 8) * HID + ct * 16 + n;
#pragma unroll
  for (int r = 0; r < 8; ++r) yp[r * HID] = acc[r];
}

// ---------- per-edge attention score + segment max (wave per edge) ----------
__global__ void __launch_bounds__(256)
edge_score(const float* __restrict__ XL, const float* __restrict__ XR,
           const int* __restrict__ src, const int* __restrict__ dst,
           const float* __restrict__ att, float* __restrict__ score,
           unsigned* __restrict__ smax, int E) {
  int lane = threadIdx.x & 31;
  int e = blockIdx.x * 8 + (threadIdx.x >> 5);
  if (e >= E) return;
  int s = src[e], d = dst[e];
  float4 xl = *(const float4*)(XL + (size_t)s * HID + lane * 4);
  float4 xr = *(const float4*)(XR + (size_t)d * HID + lane * 4);
  float4 at = *(const float4*)(att + lane * 4);
  float m0 = xl.x + xr.x, m1 = xl.y + xr.y, m2 = xl.z + xr.z, m3 = xl.w + xr.w;
  float p = at.x * (m0 > 0.f ? m0 : SLOPE * m0)
          + at.y * (m1 > 0.f ? m1 : SLOPE * m1)
          + at.z * (m2 > 0.f ? m2 : SLOPE * m2)
          + at.w * (m3 > 0.f ? m3 : SLOPE * m3);
  p += __shfl_xor(p, 1, 8);
  p += __shfl_xor(p, 2, 8);
  p += __shfl_xor(p, 4, 8);
  if ((lane & 7) == 0) {
    int h = lane >> 3;
    score[(size_t)e * NH + h] = p;
    atomicMax(smax + (size_t)d * NH + h, f2ord(p));
  }
}

// ---------- exp(score - max) + segment sum ----------
__global__ void __launch_bounds__(256)
edge_expden(float* __restrict__ score, const unsigned* __restrict__ smax,
            float* __restrict__ den, const int* __restrict__ dst, int E) {
  int i = blockIdx.x * blockDim.x + threadIdx.x;
  if (i >= E * NH) return;
  int e = i >> 2, h = i & 3;
  int d = dst[e];
  float ex = __expf(score[i] - ord2f(smax[(size_t)d * NH + h]));
  score[i] = ex;
  atomAddF(den + (size_t)d * NH + h, ex);
}

// ---------- alpha-weighted message scatter (wave per edge, f32 atomics) ----------
__global__ void __launch_bounds__(256)
edge_scatter(const float* __restrict__ XL, const float* __restrict__ ex,
             const float* __restrict__ den, const int* __restrict__ src,
             const int* __restrict__ dst, float* __restrict__ out, int E) {
  int lane = threadIdx.x & 31;
  int e = blockIdx.x * 8 + (threadIdx.x >> 5);
  if (e >= E) return;
  int s = src[e], d = dst[e];
  int h = lane >> 3;
  float alpha = ex[(size_t)e * NH + h] / den[(size_t)d * NH + h];
  float4 xl = *(const float4*)(XL + (size_t)s * HID + lane * 4);
  float* o = out + (size_t)d * HID + lane * 4;
  atomAddF(o + 0, xl.x * alpha);
  atomAddF(o + 1, xl.y * alpha);
  atomAddF(o + 2, xl.z * alpha);
  atomAddF(o + 3, xl.w * alpha);
}

// ---------- residual + LayerNorm (wave per row); refresh f16 copy ----------
__global__ void __launch_bounds__(256)
ln_residual(float* __restrict__ X, const float* __restrict__ agg,
            const float* __restrict__ gbias, const float* __restrict__ g,
            const float* __restrict__ b, _Float16* __restrict__ X16, int nrows) {
  int lane = threadIdx.x & 31;
  int row = blockIdx.x * 8 + (threadIdx.x >> 5);
  if (row >= nrows) return;
  size_t base = (size_t)row * HID + lane * 4;
  float4 xv = *(const float4*)(X + base);
  float4 av = *(const float4*)(agg + base);
  float4 gb = *(const float4*)(gbias + lane * 4);
  float v0 = xv.x + av.x + gb.x, v1 = xv.y + av.y + gb.y;
  float v2 = xv.z + av.z + gb.z, v3 = xv.w + av.w + gb.w;
  float sum = v0 + v1 + v2 + v3;
  float sq = v0 * v0 + v1 * v1 + v2 * v2 + v3 * v3;
#pragma unroll
  for (int off = 1; off < 32; off <<= 1) {
    sum += __shfl_xor(sum, off, 32);
    sq  += __shfl_xor(sq, off, 32);
  }
  float mu = sum * (1.f / HID);
  float var = sq * (1.f / HID) - mu * mu;
  float rs = rsqrtf(var + LNEPS);
  float4 gg = *(const float4*)(g + lane * 4);
  float4 bb = *(const float4*)(b + lane * 4);
  float y0 = (v0 - mu) * rs * gg.x + bb.x;
  float y1 = (v1 - mu) * rs * gg.y + bb.y;
  float y2 = (v2 - mu) * rs * gg.z + bb.z;
  float y3 = (v3 - mu) * rs * gg.w + bb.w;
  *(float4*)(X + base) = make_float4(y0, y1, y2, y3);
  _Float16* h16 = X16 + base;
  h16[0] = (_Float16)y0; h16[1] = (_Float16)y1;
  h16[2] = (_Float16)y2; h16[3] = (_Float16)y3;
}

// ---------- final column mean over NCLS rows ----------
__global__ void colmean(const float* __restrict__ XC, float* __restrict__ out, int nrows) {
  int c = threadIdx.x;  // 128 threads
  float acc = 0.f;
  for (int r = blockIdx.x; r < nrows; r += gridDim.x) acc += XC[(size_t)r * HID + c];
  atomAddF(out + c, acc * (1.f / NCLS));
}

extern "C" void kernel_launch(void* const* d_in, const int* in_sizes, int n_in,
                              void* d_out, int out_size, void* d_ws, size_t ws_size,
                              hipStream_t stream) {
  (void)in_sizes; (void)n_in; (void)out_size; (void)ws_size;
  const float* x_node       = (const float*)d_in[0];
  const float* x_class      = (const float*)d_in[1];
  const int*   member_src   = (const int*)d_in[2];
  const int*   member_dst   = (const int*)d_in[3];
  const int*   contains_src = (const int*)d_in[4];
  const int*   contains_dst = (const int*)d_in[5];
  const float* npw = (const float*)d_in[6];
  const float* npb = (const float*)d_in[7];
  const float* cpw = (const float*)d_in[8];
  const float* cpb = (const float*)d_in[9];
  const float* n2c_wl   = (const float*)d_in[10];
  const float* n2c_bl   = (const float*)d_in[11];
  const float* n2c_wr   = (const float*)d_in[12];
  const float* n2c_br   = (const float*)d_in[13];
  const float* n2c_att  = (const float*)d_in[14];
  const float* n2c_bias = (const float*)d_in[15];
  const float* c2n_wl   = (const float*)d_in[16];
  const float* c2n_bl   = (const float*)d_in[17];
  const float* c2n_wr   = (const float*)d_in[18];
  const float* c2n_br   = (const float*)d_in[19];
  const float* c2n_att  = (const float*)d_in[20];
  const float* c2n_bias = (const float*)d_in[21];
  const float* ln_cg = (const float*)d_in[22];
  const float* ln_cb = (const float*)d_in[23];
  const float* ln_ng = (const float*)d_in[24];
  const float* ln_nb = (const float*)d_in[25];

  // bump allocator over workspace
  char* base = (char*)d_ws;
  size_t off = 0;
  auto alloc = [&](size_t bytes) -> char* {
    char* p = base + off;
    off = (off + bytes + 255) & ~(size_t)255;
    return p;
  };
  float*    XN    = (float*)alloc((size_t)NNODE * HID * 4);
  float*    XC    = (float*)alloc((size_t)NCLS  * HID * 4);
  _Float16* XN16  = (_Float16*)alloc((size_t)NNODE * HID * 2);
  _Float16* XC16  = (_Float16*)alloc((size_t)NCLS  * HID * 2);
  float*    XL    = (float*)alloc((size_t)NNODE * HID * 4);
  float*    XR    = (float*)alloc((size_t)NNODE * HID * 4);
  float*    OUTB  = (float*)alloc((size_t)NNODE * HID * 4);
  _Float16* WPK   = (_Float16*)alloc((size_t)2 * 128 * 128 * 2);
  float*    SCORE = (float*)alloc((size_t)NEDGE * NH * 4);
  unsigned* SMAX  = (unsigned*)alloc((size_t)NNODE * NH * 4);
  float*    DEN   = (float*)alloc((size_t)NNODE * NH * 4);

  // input projections (fp32 VALU; tiny K)
  proj_kernel<NDIM><<<(NNODE * HID + 255) / 256, 256, 0, stream>>>(x_node, npw, npb, XN, XN16, NNODE);
  proj_kernel<CDIM><<<(NCLS  * HID + 255) / 256, 256, 0, stream>>>(x_class, cpw, cpb, XC, XC16, NCLS);

  for (int l = 0; l < 4; ++l) {
    // ---------- n2c: node -> class ----------
    pack_w<<<8, 256, 0, stream>>>(n2c_wl + (size_t)l * HID * HID, n2c_wr + (size_t)l * HID * HID, WPK);
    gemm128_wmma<<<NNODE / 16, 256, 0, stream>>>(XN16, WPK,         n2c_bl + l * HID, XL, NNODE);
    gemm128_wmma<<<NCLS  / 16, 256, 0, stream>>>(XC16, WPK + 16384, n2c_br + l * HID, XR, NCLS);
    init_softmax_bufs<<<((long)NCLS * NH + 255) / 256, 256, 0, stream>>>(SMAX, DEN, (long)NCLS * NH);
    fill_u32<<<((long)NCLS * HID + 255) / 256, 256, 0, stream>>>((unsigned*)OUTB, 0u, (long)NCLS * HID);
    edge_score<<<NEDGE / 8, 256, 0, stream>>>(XL, XR, member_src, member_dst,
                                              n2c_att + l * HID, SCORE, SMAX, NEDGE);
    edge_expden<<<((long)NEDGE * NH + 255) / 256, 256, 0, stream>>>(SCORE, SMAX, DEN, member_dst, NEDGE);
    edge_scatter<<<NEDGE / 8, 256, 0, stream>>>(XL, SCORE, DEN, member_src, member_dst, OUTB, NEDGE);
    ln_residual<<<(NCLS + 7) / 8, 256, 0, stream>>>(XC, OUTB, n2c_bias + l * HID,
                                                    ln_cg + l * HID, ln_cb + l * HID, XC16, NCLS);

    // ---------- c2n: class -> node ----------
    pack_w<<<8, 256, 0, stream>>>(c2n_wl + (size_t)l * HID * HID, c2n_wr + (size_t)l * HID * HID, WPK);
    gemm128_wmma<<<NCLS  / 16, 256, 0, stream>>>(XC16, WPK,         c2n_bl + l * HID, XL, NCLS);
    gemm128_wmma<<<NNODE / 16, 256, 0, stream>>>(XN16, WPK + 16384, c2n_br + l * HID, XR, NNODE);
    init_softmax_bufs<<<((long)NNODE * NH + 255) / 256, 256, 0, stream>>>(SMAX, DEN, (long)NNODE * NH);
    fill_u32<<<((long)NNODE * HID + 255) / 256, 256, 0, stream>>>((unsigned*)OUTB, 0u, (long)NNODE * HID);
    edge_score<<<NEDGE / 8, 256, 0, stream>>>(XL, XR, contains_src, contains_dst,
                                              c2n_att + l * HID, SCORE, SMAX, NEDGE);
    edge_expden<<<((long)NEDGE * NH + 255) / 256, 256, 0, stream>>>(SCORE, SMAX, DEN, contains_dst, NEDGE);
    edge_scatter<<<NEDGE / 8, 256, 0, stream>>>(XL, SCORE, DEN, contains_src, contains_dst, OUTB, NEDGE);
    ln_residual<<<(NNODE + 7) / 8, 256, 0, stream>>>(XN, OUTB, c2n_bias + l * HID,
                                                     ln_ng + l * HID, ln_nb + l * HID, XN16, NNODE);
  }

  // final mean over class rows -> [1, HID]
  fill_u32<<<1, 128, 0, stream>>>((unsigned*)d_out, 0u, 128);
  colmean<<<256, 128, 0, stream>>>(XC, (float*)d_out, NCLS);
}
